// SparseMoE_7911329759614
// MI455X (gfx1250) — matrix operand order repaired and checked by reading, not compiled, not run on hardware
//
#include <hip/hip_runtime.h>
#include <hip/hip_bf16.h>
#include <math.h>

#define B_ 8
#define N_ 2048
#define D_ 1024
#define E_ 8
#define F_ 4096
#define H_ 1024
#define T_ (B_ * N_)               // 16384 tokens

#define TILE_M 32                  // tokens per expert tile (2 M-subtiles)
#define THREADS 512                // 16 waves: 2 M-groups x 8 N-groups
#define FC 512                     // F-chunk staged in LDS
#define FPAD (FC + 8)              // row stride 520*2=1040B, 16B-aligned
#define TILES_PER_E (T_ / TILE_M)  // 512

#define KT1 (D_ / 32)   // 32   k-tiles of W1
#define NT1 (F_ / 16)   // 256  n-tiles of W1
#define KT2 (F_ / 32)   // 128  k-tiles of W2
#define NT2 (H_ / 16)   // 64   n-tiles of W2

typedef __attribute__((ext_vector_type(16))) __bf16        v16bf;
typedef __attribute__((ext_vector_type(8)))  float         v8f;
typedef __attribute__((ext_vector_type(4)))  unsigned int  v4u;

union ABFrag { v16bf v; unsigned short u[16]; v4u q[2]; };

__device__ __forceinline__ unsigned short f2bf(float f) {
  union { float f; unsigned u; } v; v.f = f;
  unsigned r = v.u + 0x7FFFu + ((v.u >> 16) & 1u);   // round-to-nearest-even
  return (unsigned short)(r >> 16);
}

// ---------------- Pre-pass: x (f32) -> bf16 row-major ----------------------
__global__ __launch_bounds__(256) void pack_x_kernel(
    const float* __restrict__ x, unsigned short* __restrict__ xb) {
  size_t i = ((size_t)blockIdx.x * blockDim.x + threadIdx.x) * 8;
  ABFrag o;  // use first 8 shorts
#pragma unroll
  for (int j = 0; j < 8; ++j) o.u[j] = f2bf(x[i + j]);
  *(v4u*)(xb + i) = o.q[0];
}

// ---- Pre-pass: W[K,N] f32 -> bf16 packed in WMMA B-fragment order ---------
// Layout: pW[((e*KT + kt)*NT + nt)*32 + lane][16], element j = W[e][kt*32 +
// (j<8 ? j : j+8) + 8*(lane>>4)][nt*16 + (lane&15)]  (ISA wave32 16-bit B layout).
__global__ __launch_bounds__(256) void pack_w_kernel(
    const float* __restrict__ W, unsigned short* __restrict__ pW,
    int K, int N, int KT, int NT) {
  size_t tid  = (size_t)blockIdx.x * blockDim.x + threadIdx.x;
  int    lane = (int)(tid & 31);
  size_t frag = tid >> 5;
  int nt = (int)(frag % NT);
  size_t r   = frag / NT;
  int kt = (int)(r % KT);
  int e  = (int)(r / KT);
  const int half = lane >> 4, lm = lane & 15;
  const float* We = W + (size_t)e * K * N;
  ABFrag o;
#pragma unroll
  for (int j = 0; j < 16; ++j) {
    int k = kt * 32 + (j < 8 ? j : j + 8) + 8 * half;
    o.u[j] = f2bf(We[(size_t)k * N + nt * 16 + lm]);
  }
  unsigned short* dst = pW + tid * 16;
  *(v4u*)(dst)     = o.q[0];
  *(v4u*)(dst + 8) = o.q[1];
}

// ---------------- Router: logits -> top-2 -> softmax -> per-expert lists ----
__global__ __launch_bounds__(256) void moe_router_kernel(
    const float* __restrict__ x, const float* __restrict__ Wg,
    const float* __restrict__ bg, int* __restrict__ counts,
    int* __restrict__ tokList, float* __restrict__ gateList) {
  const int wave = threadIdx.x >> 5;
  const int lane = threadIdx.x & 31;
  const int t = blockIdx.x * 8 + wave;
  if (t >= T_) return;

  float acc[E_];
#pragma unroll
  for (int e = 0; e < E_; ++e) acc[e] = 0.f;

  const float* xr = x + (size_t)t * D_;
  for (int d = lane; d < D_; d += 32) {
    float xv = xr[d];
    const float* wr = Wg + (size_t)d * E_;
#pragma unroll
    for (int e = 0; e < E_; ++e) acc[e] += xv * wr[e];
  }
#pragma unroll
  for (int off = 16; off >= 1; off >>= 1) {
#pragma unroll
    for (int e = 0; e < E_; ++e) acc[e] += __shfl_xor(acc[e], off, 32);
  }
  if (lane == 0) {
#pragma unroll
    for (int e = 0; e < E_; ++e) acc[e] += bg[e];
    int e1 = 0; float v1 = acc[0];
#pragma unroll
    for (int e = 1; e < E_; ++e) if (acc[e] > v1) { v1 = acc[e]; e1 = e; }
    int e2 = -1; float v2 = -3.4e38f;
#pragma unroll
    for (int e = 0; e < E_; ++e) if (e != e1 && acc[e] > v2) { v2 = acc[e]; e2 = e; }
    float g1 = 1.f / (1.f + __expf(v2 - v1));
    float g2 = 1.f - g1;
    int p1 = atomicAdd(&counts[e1], 1);
    tokList[e1 * T_ + p1] = t; gateList[e1 * T_ + p1] = g1;
    int p2 = atomicAdd(&counts[e2], 1);
    tokList[e2 * T_ + p2] = t; gateList[e2 * T_ + p2] = g2;
  }
}

// ---------------- Fused expert: y = relu(X*W1+b1)*W2+b2, gate-weighted ------
// 512 threads = 16 waves = 2 M-groups (16 rows each) x 8 N-groups.
__global__ __launch_bounds__(THREADS) void moe_expert_kernel(
    const unsigned short* __restrict__ xb,
    const unsigned short* __restrict__ pW1, const float* __restrict__ b1,
    const unsigned short* __restrict__ pW2, const float* __restrict__ b2,
    const int* __restrict__ counts, const int* __restrict__ tokList,
    const float* __restrict__ gateList, float* __restrict__ out) {
  __shared__ __align__(16) unsigned short hTile[TILE_M][FPAD];  // ~33 KB
  __shared__ int   sTok[TILE_M];
  __shared__ float sGate[TILE_M];

  const int e     = blockIdx.x / TILES_PER_E;
  const int tile  = blockIdx.x % TILES_PER_E;
  const int cnt   = counts[e];
  const int start = tile * TILE_M;
  if (start >= cnt) return;                        // uniform early-exit

  if (threadIdx.x < TILE_M) {
    int i = start + threadIdx.x;
    if (i < cnt) {
      sTok[threadIdx.x]  = tokList[e * T_ + i];
      sGate[threadIdx.x] = gateList[e * T_ + i];
    } else {
      sTok[threadIdx.x]  = 0;                      // pad row: gate 0 => no-op
      sGate[threadIdx.x] = 0.f;
    }
  }
  __syncthreads();

  const int wave = threadIdx.x >> 5;
  const int lane = threadIdx.x & 31;
  const int half = lane >> 4;
  const int lm   = lane & 15;
  const int mg   = wave >> 3;                      // M-group: rows [mg*16, +16)
  const int ng   = wave & 7;                       // N-group

  const unsigned short* xrow = xb + (size_t)sTok[mg * 16 + lm] * D_;
  const float* b1e = b1 + (size_t)e * F_;
  const float* b2e = b2 + (size_t)e * H_;

  v8f acc[8];                                      // y tiles: H cols [ng*128, +128)
#pragma unroll
  for (int i = 0; i < 8; ++i) acc[i] = (v8f){0.f,0.f,0.f,0.f,0.f,0.f,0.f,0.f};

  for (int c = 0; c < F_ / FC; ++c) {
    // ---- phase 1: h_chunk[32][FC] = relu(X * W1[:, cFC:cFC+FC] + b1)
    v8f hc[4];
#pragma unroll
    for (int i = 0; i < 4; ++i) hc[i] = (v8f){0.f,0.f,0.f,0.f,0.f,0.f,0.f,0.f};

    for (int k0 = 0; k0 < D_; k0 += 32) {
      ABFrag a;                                    // ISA A layout: two 16B runs
      a.q[0] = *(const v4u*)(xrow + k0 + 8 * half);
      a.q[1] = *(const v4u*)(xrow + k0 + 16 + 8 * half);
      const int kt = k0 >> 5;
      const unsigned short* bbase =
          pW1 + ((((size_t)e * KT1 + kt) * NT1 + (c * (FC / 16) + ng * 4)) * 32 + lane) * 16;
      __builtin_prefetch(bbase + (size_t)NT1 * 32 * 16, 0, 1);  // next k-tile
#pragma unroll
      for (int st = 0; st < 4; ++st) {
        ABFrag b = *(const ABFrag*)(bbase + (size_t)st * 32 * 16);
        hc[st] = __builtin_amdgcn_wmma_f32_16x16x32_bf16(
            false, a.v, false, b.v, (short)0, hc[st], false, false);
      }
    }
#pragma unroll
    for (int st = 0; st < 4; ++st) {
      const int col0 = (ng * 4 + st) * 16;         // chunk-local column base
      const float bias = b1e[c * FC + col0 + lm];
#pragma unroll
      for (int r = 0; r < 8; ++r) {
        int M = mg * 16 + r + 8 * half;
        float v = hc[st][r] + bias;
        v = v > 0.f ? v : 0.f;
        hTile[M][col0 + lm] = f2bf(v);
      }
    }
    __syncthreads();

    // ---- phase 2: acc += h_chunk * W2[cFC:cFC+FC, :]
    for (int kc = 0; kc < FC; kc += 32) {
      ABFrag a;                                    // A from LDS: two 16B ds loads
      a.q[0] = *(const v4u*)&hTile[mg * 16 + lm][kc + 8 * half];
      a.q[1] = *(const v4u*)&hTile[mg * 16 + lm][kc + 16 + 8 * half];
      const int kt2 = (c * FC + kc) >> 5;
      const unsigned short* bbase2 =
          pW2 + ((((size_t)e * KT2 + kt2) * NT2 + ng * 8) * 32 + lane) * 16;
      __builtin_prefetch(bbase2 + (size_t)NT2 * 32 * 16, 0, 1); // next k-tile
#pragma unroll
      for (int yt = 0; yt < 8; ++yt) {
        ABFrag b = *(const ABFrag*)(bbase2 + (size_t)yt * 32 * 16);
        acc[yt] = __builtin_amdgcn_wmma_f32_16x16x32_bf16(
            false, a.v, false, b.v, (short)0, acc[yt], false, false);
      }
    }
    __syncthreads();
  }

  // ---- epilogue: out[batch(t), :] += gate_t * (y + b2)
#pragma unroll
  for (int yt = 0; yt < 8; ++yt) {
    const int n = (ng * 8 + yt) * 16 + lm;
    const float bias2 = b2e[n];
#pragma unroll
    for (int r = 0; r < 8; ++r) {
      int M = mg * 16 + r + 8 * half;
      float g = sGate[M];
      int  t  = sTok[M];
      int  bi = t / N_;
      atomicAdd(&out[(size_t)bi * H_ + n], g * (acc[yt][r] + bias2));
    }
  }
}

// ---------------------------------------------------------------------------
extern "C" void kernel_launch(void* const* d_in, const int* in_sizes, int n_in,
                              void* d_out, int out_size, void* d_ws, size_t ws_size,
                              hipStream_t stream) {
  (void)in_sizes; (void)n_in; (void)out_size; (void)ws_size;
  const float* x  = (const float*)d_in[0];
  const float* Wg = (const float*)d_in[1];
  const float* bg = (const float*)d_in[2];
  const float* W1 = (const float*)d_in[3];
  const float* b1 = (const float*)d_in[4];
  const float* W2 = (const float*)d_in[5];
  const float* b2 = (const float*)d_in[6];
  float* out = (float*)d_out;

  char* ws = (char*)d_ws;
  size_t o = 0;
  int* counts = (int*)(ws + o);               o += 4096;
  int* tokList = (int*)(ws + o);              o += (size_t)E_ * T_ * sizeof(int);
  float* gateList = (float*)(ws + o);         o += (size_t)E_ * T_ * sizeof(float);
  o = (o + 255) & ~(size_t)255;
  unsigned short* xb  = (unsigned short*)(ws + o); o += (size_t)T_ * D_ * 2;      // 32 MB
  unsigned short* pW1 = (unsigned short*)(ws + o); o += (size_t)E_ * D_ * F_ * 2; // 64 MB
  unsigned short* pW2 = (unsigned short*)(ws + o); o += (size_t)E_ * F_ * H_ * 2; // 64 MB

  hipMemsetAsync(counts, 0, 4096, stream);
  hipMemsetAsync(out, 0, sizeof(float) * B_ * H_, stream);

  // pre-passes (independent of router)
  pack_x_kernel<<<(int)(((size_t)T_ * D_ / 8) / 256), 256, 0, stream>>>(x, xb);
  pack_w_kernel<<<(int)(((size_t)E_ * KT1 * NT1 * 32) / 256), 256, 0, stream>>>(
      W1, pW1, D_, F_, KT1, NT1);
  pack_w_kernel<<<(int)(((size_t)E_ * KT2 * NT2 * 32) / 256), 256, 0, stream>>>(
      W2, pW2, F_, H_, KT2, NT2);

  moe_router_kernel<<<T_ / 8, 256, 0, stream>>>(x, Wg, bg, counts, tokList, gateList);

  moe_expert_kernel<<<E_ * TILES_PER_E, THREADS, 0, stream>>>(
      xb, pW1, b1, pW2, b2, counts, tokList, gateList, out);
}
